// RNAPocketEncoder_25022479466499
// MI455X (gfx1250) — matrix operand order repaired and checked by reading, not compiled, not run on hardware
//
#include <hip/hip_runtime.h>
#include <hip/hip_bf16.h>
#include <math.h>

// ---------------------------------------------------------------------------
// RNAPocketEncoder forward for MI455X (gfx1250), wave32, WMMA bf16 path.
//
// d_in leaf order (setup_inputs dict insertion order):
//  0 x(4096,64) 1 pos(4096,3) 2 edge_index(2,16384) 3 batch(4096) 4 w_emb(64,32)
//  layer i (i=0,1) base 5+12i: sc0(32,56) sc1(16,16) sc2(8,8)
//                              lin0(32,32) lin1(16,16) lin2(8,8)
//                              w0(8,64) b0(64) w1(64,64) b1(64) w2(64,4800) b2(4800)
//  29..34 pool: w0(32,128) b0 w1(128,128) b1 w2(128,1) b2
//  35..38 out : w(32,512) b g be
// ---------------------------------------------------------------------------

#define NNODE 4096
#define NEDGE 16384
#define NGRAPH 16
#define FIN   120     // 32x0e + 16x1e + 8x2e
#define FMID  144     // 56x0e + 16x1e + 8x2e
#define WNUM  4800
#define TLEN  592     // T tensor floats per edge
#define W3LEN 363
#define NTILE 300     // 4800/16

typedef __attribute__((ext_vector_type(16))) __bf16 v16bf;
typedef __attribute__((ext_vector_type(8)))  float  v8f;
typedef int v4i_t __attribute__((__vector_size__(16)));

#if __has_builtin(__builtin_amdgcn_global_load_async_to_lds_b128) && \
    __has_builtin(__builtin_amdgcn_s_wait_asynccnt)
#define HAVE_ASYNC_LDS 1
typedef __attribute__((address_space(1))) v4i_t* gv4i_p;
typedef __attribute__((address_space(3))) v4i_t* lv4i_p;
#endif

// ---- instruction metadata (compile-time) ----------------------------------
__constant__ int   c_Ioff[18] = {0,1024,1536,1792,2304,2560,2816,3328,3584,3712,
                                 3840,4096,4160,4288,4544,4672,4736,4800};
__constant__ int   c_Imulo[17]= {32,16,8,16,8,16,32,16,8,8,16,8,16,32,16,8,8};
__constant__ int   c_Ish[17]  = {5,4,3,4,3,4,5,4,3,3,4,3,4,5,4,3,3}; // log2(mulo)
__constant__ int   c_Ilo[17]  = {0,0,0,1,2,1,0,0,0,2,1,2,1,0,0,0,2};
__constant__ int   c_Iio[17]  = {0,1,2,3,4,3,0,1,2,4,3,4,3,0,1,2,4};
__constant__ int   c_Itk[17]  = {0,0,0,1,2,3,4,4,4,5,6,7,8,9,9,9,10};
__constant__ int   c_OutOff[5]= {0,32,48,56,104};
__constant__ float c_Alpha[5] = {0.13363062f,0.13363062f,0.13363062f,
                                 0.20412414f,0.27950850f}; // sqrt((2lo+1)/FAN)
// T-keys: (l1,l2,lo) dedup of instructions
__constant__ int c_TKl1[11]  = {0,0,0,1,1,1,1,2,2,2,2};
__constant__ int c_TKl2[11]  = {0,1,2,0,1,1,2,0,1,2,2};
__constant__ int c_TKlo[11]  = {0,1,2,1,0,2,1,2,1,0,2};
__constant__ int c_TKtoff[12]= {0,32,128,288,336,352,432,480,520,544,552,592};
__constant__ int c_TKw3[11]  = {0,1,10,35,44,53,98,143,168,213,238};
__constant__ int c_XJoff[3]  = {0,32,80};   // xj feature base by l1
__constant__ int c_SHoff[3]  = {0,1,4};     // sh base by l2

// ---- helpers --------------------------------------------------------------
__device__ __forceinline__ unsigned short f2bf(float f) {
  unsigned u = __builtin_bit_cast(unsigned, f);
  return (unsigned short)((u + 0x7FFFu + ((u >> 16) & 1u)) >> 16);
}
__device__ __forceinline__ __bf16 usbf(unsigned short u) {
  return __builtin_bit_cast(__bf16, u);
}
__device__ __forceinline__ float siluf(float x) { return x / (1.f + expf(-x)); }
__device__ __forceinline__ void atomAddLds(float* p, float v) {
  __hip_atomic_fetch_add(p, v, __ATOMIC_RELAXED, __HIP_MEMORY_SCOPE_WORKGROUP);
}
__device__ __forceinline__ void atomAddGlb(float* p, float v) {
  __hip_atomic_fetch_add(p, v, __ATOMIC_RELAXED, __HIP_MEMORY_SCOPE_AGENT);
}

// unrolled tensor-product accumulation (NK = 2*lo+1, wave-uniform)
template <int NK>
__device__ __forceinline__ void tp_accum(float* sOut, const float* sT, int halfA,
                                         const v8f& acc, float bias, float alpha,
                                         int ooff, int toff){
  #pragma unroll
  for (int v=0; v<8; v++){
    int e = v + halfA;
    float wv = alpha*(acc[v] + bias);
    #pragma unroll
    for (int k=0;k<NK;k++)
      atomAddLds(&sOut[e*FMID + ooff + k], wv * sT[e*TLEN + toff + k]);
  }
}

// ===========================================================================
// Init kernel: Wigner-3j tables (thread 0, exact CG construction) + _mom2
// activation normalization constants (all 256 threads, 400001-pt quadrature).
// ===========================================================================
struct cplx { double re, im; };
__device__ cplx cmul(cplx a, cplx b){ return {a.re*b.re-a.im*b.im, a.re*b.im+a.im*b.re}; }
__device__ double dfact(int n){ double r=1.0; for(int i=2;i<=n;i++) r*=(double)i; return r; }

__device__ double su2_cg(int j1,int m1,int j2,int m2,int j3,int m3){
  if (m3 != m1+m2) return 0.0;
  int vmin = max(-j1+j2+m3, max(-j1+m1, 0));
  int vmax = min(j2+j3+m1, min(j3-j1+j2, j3+m3));
  double c = sqrt((double)(2*j3+1)*dfact(j3+j1-j2)*dfact(j3-j1+j2)*dfact(j1+j2-j3)
                  *dfact(j3+m3)*dfact(j3-m3)
                  /(dfact(j1+j2+j3+1)*dfact(j1-m1)*dfact(j1+m1)*dfact(j2-m2)*dfact(j2+m2)));
  double s = 0.0;
  for (int v=vmin; v<=vmax; v++){
    double sgn = ((v+j2+m2)&1) ? -1.0 : 1.0;
    s += sgn*dfact(j2+j3+m1-v)*dfact(j1-m1+v)
           /(dfact(v)*dfact(j3-j1+j2-v)*dfact(j3+m3-v)*dfact(v+j1-j2-m3));
  }
  return c*s;
}
__device__ void qmat(int l, cplx* q){  // (2l+1)^2, row-major
  int n = 2*l+1;
  for (int i=0;i<n*n;i++) q[i] = {0.0,0.0};
  const double is2 = 0.70710678118654752440;
  for (int m=-l; m<0; m++){
    q[(l+m)*n + (l-m)] = {is2, 0.0};     // col l+|m|
    q[(l+m)*n + (l+m)] = {0.0, -is2};    // col l-|m|
  }
  q[l*n+l] = {1.0,0.0};
  for (int m=1; m<=l; m++){
    double sg = (m&1) ? -1.0 : 1.0;
    q[(l+m)*n + (l+m)] = {sg*is2, 0.0};
    q[(l+m)*n + (l-m)] = {0.0, sg*is2};
  }
  cplx f; switch (l&3){ case 0:f={1,0};break; case 1:f={0,-1};break;
                        case 2:f={-1,0};break; default:f={0,1}; }
  for (int i=0;i<n*n;i++) q[i] = cmul(q[i], f);
}

__global__ void k_init(float* w3, float* cst){
  int tid = threadIdx.x;
  // ---- mom2 constants (parallel quadrature, deterministic LDS tree) ----
  __shared__ double rS[256], rG[256];
  double sS=0.0, sG=0.0;
  const double dx = 24.0/400000.0, inv_s2pi = 0.39894228040143267794;
  for (int i=tid; i<=400000; i+=256){
    double x = -12.0 + dx*(double)i;
    double w = exp(-0.5*x*x)*inv_s2pi;
    double sg = 1.0/(1.0+exp(-x));
    double si = x*sg;
    sS += si*si*w; sG += sg*sg*w;
  }
  rS[tid]=sS; rG[tid]=sG; __syncthreads();
  for (int s=128; s>0; s>>=1){ if (tid<s){ rS[tid]+=rS[tid+s]; rG[tid]+=rG[tid+s]; } __syncthreads(); }
  if (tid==0){ cst[0]=(float)(1.0/sqrt(rS[0]*dx)); cst[1]=(float)(1.0/sqrt(rG[0]*dx)); }
  // ---- Wigner 3j tables (single thread; tiny) ----
  if (tid==0){
    cplx q1[25],q2[25],q3[25]; double c[125], d[125];
    for (int key=0; key<11; key++){
      int l1=c_TKl1[key], l2=c_TKl2[key], l3=c_TKlo[key];
      int n1=2*l1+1, n2=2*l2+1, n3=2*l3+1;
      for (int i=0;i<n1;i++) for (int k=0;k<n2;k++) for (int n=0;n<n3;n++)
        c[(i*n2+k)*n3+n] = su2_cg(l1,i-l1,l2,k-l2,l3,n-l3);
      qmat(l1,q1); qmat(l2,q2); qmat(l3,q3);
      double nrm2 = 0.0;
      for (int j=0;j<n1;j++) for (int l=0;l<n2;l++) for (int m=0;m<n3;m++){
        cplx acc = {0.0,0.0};
        for (int i=0;i<n1;i++) for (int k=0;k<n2;k++) for (int n=0;n<n3;n++){
          double cv = c[(i*n2+k)*n3+n];
          if (cv == 0.0) continue;
          cplx t = cmul(q1[i*n1+j], q2[k*n2+l]);
          cplx qc = {q3[n*n3+m].re, -q3[n*n3+m].im};  // conj
          t = cmul(t, qc);
          acc.re += t.re*cv; acc.im += t.im*cv;
        }
        d[(j*n2+l)*n3+m] = acc.re;
        nrm2 += acc.re*acc.re;
      }
      double inv = 1.0/sqrt(nrm2);
      for (int i=0;i<n1*n2*n3;i++) w3[c_TKw3[key]+i] = (float)(d[i]*inv);
    }
  }
}

// ===========================================================================
// Misc small kernels
// ===========================================================================
__global__ void k_zero(float* p, int n){
  int i = blockIdx.x*256 + threadIdx.x; if (i<n) p[i]=0.f;
}

__global__ void k_embed(const float* __restrict__ x, const float* __restrict__ wemb,
                        float* __restrict__ h){
  int i = blockIdx.x*256 + threadIdx.x;
  if (i >= NNODE*FIN) return;
  int n = i/FIN, f = i - n*FIN;
  float v = 0.f;
  if (f < 32){
    const float* xr = x + n*64;
    for (int c=0;c<64;c++) v += xr[c]*wemb[c*32+f];
    v *= 0.125f;                         // 1/sqrt(64)
  }
  h[i] = v;
}

__global__ void k_w2t(const float* __restrict__ w2, unsigned short* __restrict__ w2t){
  int i = blockIdx.x*256 + threadIdx.x;
  if (i >= WNUM*64) return;
  int col = i >> 6, c = i & 63;
  w2t[i] = f2bf(w2[c*WNUM + col]);       // [col][c] <- w2[c][col]
}

// node linears: x_self (sc) and h_lin (lin)
__global__ void k_nodelin(const float* __restrict__ h,
                          const float* sc0,const float* sc1,const float* sc2,
                          const float* ln0,const float* ln1,const float* ln2,
                          float* __restrict__ xself, float* __restrict__ hlin){
  __shared__ float hh[FIN];
  int n = blockIdx.x, tid = threadIdx.x;
  for (int i=tid;i<FIN;i+=128) hh[i]=h[n*FIN+i];
  __syncthreads();
  const float i32=0.17677669529663688f, i16=0.25f, i8=0.35355339059327373f;
  for (int f=tid; f<264; f+=128){
    float a = 0.f;
    if (f < 144){                        // x_self (56x0e+16x1e+8x2e)
      if (f < 56){ for (int u=0;u<32;u++) a += sc0[u*56+f]*hh[u]; a*=i32; }
      else if (f < 104){ int t=f-56, v=t/3, m=t-3*v;
        for (int u=0;u<16;u++) a += sc1[u*16+v]*hh[32+3*u+m]; a*=i16; }
      else { int t=f-104, v=t/5, m=t-5*v;
        for (int u=0;u<8;u++)  a += sc2[u*8+v]*hh[80+5*u+m];  a*=i8; }
      xself[n*FMID+f]=a;
    } else {                             // h_lin (32x0e+16x1e+8x2e)
      int g=f-144;
      if (g < 32){ for (int u=0;u<32;u++) a += ln0[u*32+g]*hh[u]; a*=i32; }
      else if (g < 80){ int t=g-32, v=t/3, m=t-3*v;
        for (int u=0;u<16;u++) a += ln1[u*16+v]*hh[32+3*u+m]; a*=i16; }
      else { int t=g-80, v=t/5, m=t-5*v;
        for (int u=0;u<8;u++)  a += ln2[u*8+v]*hh[80+5*u+m];  a*=i8; }
      hlin[n*FIN+g]=a;
    }
  }
}

// ===========================================================================
// Edge kernel: geometry + radial MLP + WMMA(bf16) GEMM fused with tensor
// product, LDS message accumulation, atomic scatter to agg.
// 1 block = 16 edges, 256 threads = 8 waves, 300 N-tiles of 16 columns.
// xj gather uses GLOBAL_LOAD_ASYNC_TO_LDS_B128 (ASYNCcnt) when available.
// ===========================================================================
__global__ __launch_bounds__(256)
void k_edge(const float* __restrict__ pos, const int* __restrict__ ei,
            const float* __restrict__ hlin,
            const float* w0,const float* b0,const float* w1,const float* b1,
            const unsigned short* __restrict__ w2t, const float* __restrict__ b2,
            const float* __restrict__ w3, float* __restrict__ agg){
  __shared__ float sW3[W3LEN+1];
  __shared__ float sSh[16][9];
  __shared__ float sRbf[16][8];
  __shared__ __align__(16) float sXj[16][FIN];
  __shared__ float sT[16*TLEN];
  __shared__ float sOut[16*FMID];
  __shared__ float sTmp[16*64];
  __shared__ unsigned short sH[16*64];
  __shared__ int sSrc[16], sDst[16];

  const int tid = threadIdx.x;
  const int eb  = blockIdx.x*16;

  for (int i=tid;i<W3LEN;i+=256) sW3[i]=w3[i];

  if (tid < 16){
    int e = eb + tid;
    int s = ei[e], dd = ei[NEDGE + e];
    sSrc[tid]=s; sDst[tid]=dd;
    float rx=pos[dd*3+0]-pos[s*3+0];
    float ry=pos[dd*3+1]-pos[s*3+1];
    float rz=pos[dd*3+2]-pos[s*3+2];
    float dist = sqrtf(rx*rx+ry*ry+rz*rz);
    float d = fmaxf(dist, 1e-6f);
    float ux=rx/d, uy=ry/d, uz=rz/d;
    const float s3=1.7320508075688772f, s5=2.2360679774997896f, s15=3.8729833462074170f;
    sSh[tid][0]=1.f;
    sSh[tid][1]=s3*uy; sSh[tid][2]=s3*uz; sSh[tid][3]=s3*ux;
    sSh[tid][4]=s15*ux*uy; sSh[tid][5]=s15*uy*uz;
    sSh[tid][6]=0.5f*s5*(3.f*uz*uz-1.f);
    sSh[tid][7]=s15*ux*uz; sSh[tid][8]=0.5f*s15*(ux*ux-uy*uy);
    float uu=d/6.0f, fc=0.f;
    if (uu < 1.f){
      float u2=uu*uu, u4=u2*u2, u6=u4*u2, u7=u6*uu, u8=u7*uu;
      fc = 1.f - 28.f*u6 + 48.f*u7 - 21.f*u8;
    }
    const float pi = 3.14159265358979323846f;
    for (int i=0;i<8;i++){
      float fr = pi*(float)(i+1)/6.0f;
      sRbf[tid][i] = (2.f/6.f)*sinf(fr*d)/d*fc;
    }
  }
  __syncthreads();

  // gather xj = h_lin[src]: 16 edges x 30 16-byte chunks
#ifdef HAVE_ASYNC_LDS
  for (int c=tid; c<480; c+=256){
    int e = c/30, q = c - e*30;
    const float* g = &hlin[sSrc[e]*FIN + q*4];
    __builtin_amdgcn_global_load_async_to_lds_b128(
        (gv4i_p)(void*)g,
        (lv4i_p)&sXj[e][q*4],
        0, 0);
  }
  __builtin_amdgcn_s_wait_asynccnt(0);
#else
  for (int i=tid;i<16*FIN;i+=256){ int e=i/FIN, f=i-e*FIN; sXj[e][f]=hlin[sSrc[e]*FIN+f]; }
#endif
  // radial MLP layer 0
  for (int i=tid;i<1024;i+=256){
    int e=i>>6, v=i&63;
    float a=b0[v];
    for (int c=0;c<8;c++) a += sRbf[e][c]*w0[c*64+v];
    sTmp[i]=siluf(a);
  }
  __syncthreads();

  // radial MLP layer 1 -> bf16 A matrix ; T tensors ; zero sOut
  for (int i=tid;i<1024;i+=256){
    int e=i>>6, v=i&63;
    float a=b1[v];
    for (int c=0;c<64;c++) a += sTmp[e*64+c]*w1[c*64+v];
    sH[i]=f2bf(siluf(a));
  }
  for (int i=tid;i<16*FMID;i+=256) sOut[i]=0.f;
  for (int i=tid;i<16*TLEN;i+=256){
    int e=i/TLEN, t=i-e*TLEN;
    int key=0; while (key<10 && t>=c_TKtoff[key+1]) key++;
    int loc=t-c_TKtoff[key];
    int lo=c_TKlo[key], nk=2*lo+1;
    int u=loc/nk, k=loc-u*nk;
    int l1=c_TKl1[key], l2=c_TKl2[key];
    int n1=2*l1+1, n2=2*l2+1;
    const float* W = &sW3[c_TKw3[key]];
    int xb = c_XJoff[l1] + u*n1;
    int sb = c_SHoff[l2];
    float acc=0.f;
    for (int ii=0; ii<n1; ii++){
      float xv = sXj[e][xb+ii];
      for (int jj=0; jj<n2; jj++)
        acc += W[(ii*n2+jj)*nk + k] * xv * sSh[e][sb+jj];
    }
    sT[i]=acc;
  }
  __syncthreads();

  // -------- WMMA GEMM (16 edges x 16 cols, K=64 in 2 steps) fused with TP --
  const int lane = tid & 31;
  const int wav  = tid >> 5;
  const int nn    = lane & 15;
  const int halfA = (lane < 16) ? 0 : 8;    // C/D row offset, A K-half select
  const int halfB = (lane < 16) ? 0 : 16;   // unused; B uses halfA for its K half
  const int m     = lane & 15;              // A row (edge)

  for (int tile = wav; tile < NTILE; tile += 8){
    int col0 = tile*16;
    int col  = col0 + nn;
    // prefetch next B tile for this wave
    int nc0 = (tile+8)*16;
    if (nc0 < WNUM) __builtin_prefetch(&w2t[nc0*64], 0, 1);
    // instruction for this tile (uniform: boundaries are multiples of 16)
    int ins=0; while (ins<16 && col0 >= c_Ioff[ins+1]) ins++;

    v8f acc = {};
    #pragma unroll
    for (int kk=0; kk<2; kk++){
      int kb = kk*32;
      v16bf a, b;
      #pragma unroll
      for (int j=0;j<16;j++){
        int kA = kb + halfA + ((j<8) ? j : j+8);        // ISA 16-bit A layout
        a[j] = usbf(sH[m*64 + kA]);
        int kB = kb + halfA + ((j<8) ? j : j+8);        // ISA 16-bit B layout
        b[j] = usbf(w2t[col*64 + kB]);
      }
      acc = __builtin_amdgcn_wmma_f32_16x16x32_bf16(false, a, false, b,
                                                    (short)0, acc, false, false);
    }
    float bias = b2[col];

    // tensor product consumption of this 16x16 weight tile (shift decode:
    // mulo is always a power of two -> no integer division in the hot loop)
    int local = col - c_Ioff[ins];
    int sh    = c_Ish[ins];
    int u     = local >> sh;
    int w     = local & (c_Imulo[ins]-1);
    int lo    = c_Ilo[ins];
    int nk    = 2*lo+1;
    int toff  = c_TKtoff[c_Itk[ins]] + u*nk;
    int ooff  = c_OutOff[c_Iio[ins]] + w*nk;
    float alpha = c_Alpha[c_Iio[ins]];
    if (lo == 0)      tp_accum<1>(sOut, sT, halfA, acc, bias, alpha, ooff, toff);
    else if (lo == 1) tp_accum<3>(sOut, sT, halfA, acc, bias, alpha, ooff, toff);
    else              tp_accum<5>(sOut, sT, halfA, acc, bias, alpha, ooff, toff);
  }
  __syncthreads();

  // scatter messages to global agg
  for (int i=tid;i<16*FMID;i+=256){
    int e=i/FMID, f=i-e*FMID;
    atomAddGlb(&agg[sDst[e]*FMID + f], sOut[i]);
  }
}

// gate + residual: h_out = h_in + gate(x_self + agg)
__global__ void k_gate(const float* __restrict__ hin, const float* __restrict__ xself,
                       const float* __restrict__ agg, const float* __restrict__ cst,
                       float* __restrict__ hout){
  int n = blockIdx.x*256 + threadIdx.x;
  if (n >= NNODE) return;
  const float SC = cst[0], GC = cst[1];
  const float* xs = xself + n*FMID;
  const float* ag = agg   + n*FMID;
  float g[24];
  for (int j=0;j<24;j++){ float t = xs[32+j]+ag[32+j]; g[j] = GC/(1.f+expf(-t)); }
  for (int i=0;i<32;i++){
    float t = xs[i]+ag[i];
    hout[n*FIN+i] = hin[n*FIN+i] + SC*siluf(t);
  }
  for (int v=0;v<16;v++) for (int mm=0;mm<3;mm++){
    int f = 56+3*v+mm;
    hout[n*FIN+32+3*v+mm] = hin[n*FIN+32+3*v+mm] + (xs[f]+ag[f])*g[v];
  }
  for (int v=0;v<8;v++) for (int mm=0;mm<5;mm++){
    int f = 104+5*v+mm;
    hout[n*FIN+80+5*v+mm] = hin[n*FIN+80+5*v+mm] + (xs[f]+ag[f])*g[16+v];
  }
}

// ===========================================================================
// Pooling and head
// ===========================================================================
__global__ void k_pool_logit(const float* __restrict__ h,
                             const float* pw0,const float* pb0,
                             const float* pw1,const float* pb1,
                             const float* pw2,const float* pb2,
                             float* __restrict__ logit){
  __shared__ float sS[32], a0[128], a1[128];
  int n = blockIdx.x, tid = threadIdx.x;
  if (tid < 32) sS[tid] = h[n*FIN + tid];
  __syncthreads();
  float a = pb0[tid];
  for (int c=0;c<32;c++) a += sS[c]*pw0[c*128+tid];
  a0[tid] = siluf(a);
  __syncthreads();
  float b = pb1[tid];
  for (int c=0;c<128;c++) b += a0[c]*pw1[c*128+tid];
  a1[tid] = siluf(b);
  __syncthreads();
  if (tid == 0){
    float l = pb2[0];
    for (int c=0;c<128;c++) l += a1[c]*pw2[c];
    logit[n] = l;
  }
}

__device__ __forceinline__ unsigned fenc(float f){
  unsigned u = __builtin_bit_cast(unsigned, f);
  return (u >> 31) ? ~u : (u | 0x80000000u);
}
__device__ __forceinline__ float fdec(unsigned k){
  unsigned u = (k & 0x80000000u) ? (k & 0x7FFFFFFFu) : ~k;
  return __builtin_bit_cast(float, u);
}

__global__ void k_segmax(const float* __restrict__ logit, const int* __restrict__ batch,
                         unsigned* __restrict__ umax){
  int n = blockIdx.x*256 + threadIdx.x;
  if (n >= NNODE) return;
  atomicMax(&umax[batch[n]], fenc(logit[n]));
}
__global__ void k_expsum(const float* __restrict__ logit, const int* __restrict__ batch,
                         const unsigned* __restrict__ umax,
                         float* __restrict__ exv, float* __restrict__ zsum){
  int n = blockIdx.x*256 + threadIdx.x;
  if (n >= NNODE) return;
  int g = batch[n];
  float e = expf(logit[n] - fdec(umax[g]));
  exv[n] = e;
  atomAddGlb(&zsum[g], e);
}
__global__ void k_pooled(const float* __restrict__ exv, const float* __restrict__ zsum,
                         const int* __restrict__ batch, const float* __restrict__ h,
                         float* __restrict__ pooled){
  int n = blockIdx.x*256 + threadIdx.x;
  if (n >= NNODE) return;
  int g = batch[n];
  float att = exv[n] / fmaxf(zsum[g], 1e-12f);
  for (int c=0;c<32;c++) atomAddGlb(&pooled[g*32+c], att*h[n*FIN+c]);
}

__global__ void k_out(const float* __restrict__ pooled, const float* __restrict__ ow,
                      const float* __restrict__ ob, const float* __restrict__ og,
                      const float* __restrict__ obe, float* __restrict__ out){
  __shared__ float sp[32], sy[512], r1[256], r2[256];
  int g = blockIdx.x, tid = threadIdx.x;
  if (tid < 32) sp[tid] = pooled[g*32+tid];
  __syncthreads();
  float p1=0.f, p2=0.f;
  for (int j=tid; j<512; j+=256){
    float y = ob[j];
    for (int c=0;c<32;c++) y += sp[c]*ow[c*512+j];
    sy[j]=y; p1+=y; p2+=y*y;
  }
  r1[tid]=p1; r2[tid]=p2; __syncthreads();
  for (int s=128; s>0; s>>=1){ if (tid<s){ r1[tid]+=r1[tid+s]; r2[tid]+=r2[tid+s]; } __syncthreads(); }
  float mu  = r1[0]*(1.f/512.f);
  float var = r2[0]*(1.f/512.f) - mu*mu;
  float rs  = rsqrtf(var + 1e-5f);
  for (int j=tid; j<512; j+=256)
    out[g*512+j] = og[j]*(sy[j]-mu)*rs + obe[j];
}

// ===========================================================================
extern "C" void kernel_launch(void* const* d_in, const int* in_sizes, int n_in,
                              void* d_out, int out_size, void* d_ws, size_t ws_size,
                              hipStream_t stream){
  (void)in_sizes; (void)n_in; (void)out_size; (void)ws_size;
  const float* x    = (const float*)d_in[0];
  const float* pos  = (const float*)d_in[1];
  const int*   ei   = (const int*)d_in[2];
  const int*   batch= (const int*)d_in[3];
  const float* wemb = (const float*)d_in[4];
  const float* pw0=(const float*)d_in[29]; const float* pb0=(const float*)d_in[30];
  const float* pw1=(const float*)d_in[31]; const float* pb1=(const float*)d_in[32];
  const float* pw2=(const float*)d_in[33]; const float* pb2=(const float*)d_in[34];
  const float* ow =(const float*)d_in[35]; const float* ob =(const float*)d_in[36];
  const float* og =(const float*)d_in[37]; const float* obe=(const float*)d_in[38];

  // workspace carve-up (floats)
  float* W3   = (float*)d_ws;                 // 384 (363 used)
  float* CST  = W3 + 384;                     // 16
  float* hA   = CST + 16;                     // 4096*120
  float* hB   = hA + NNODE*FIN;
  float* xself= hB + NNODE*FIN;               // 4096*144
  float* hlin = xself + NNODE*FMID;           // 4096*120
  float* agg  = hlin + NNODE*FIN;             // 4096*144
  float* logit= agg + NNODE*FMID;             // 4096
  float* exv  = logit + NNODE;                // 4096
  unsigned* umax = (unsigned*)(exv + NNODE);  // 16
  float* zsum = (float*)(umax + 16);          // 16
  float* pooled = zsum + 16;                  // 512
  unsigned short* w2t = (unsigned short*)(pooled + 512); // 4800*64 bf16

  k_init <<<1, 256, 0, stream>>>(W3, CST);
  k_embed<<<(NNODE*FIN+255)/256, 256, 0, stream>>>(x, wemb, hA);

  for (int L=0; L<2; L++){
    int b = 5 + 12*L;
    const float* sc0=(const float*)d_in[b+0]; const float* sc1=(const float*)d_in[b+1];
    const float* sc2=(const float*)d_in[b+2];
    const float* ln0=(const float*)d_in[b+3]; const float* ln1=(const float*)d_in[b+4];
    const float* ln2=(const float*)d_in[b+5];
    const float* rw0=(const float*)d_in[b+6]; const float* rb0=(const float*)d_in[b+7];
    const float* rw1=(const float*)d_in[b+8]; const float* rb1=(const float*)d_in[b+9];
    const float* rw2=(const float*)d_in[b+10];const float* rb2=(const float*)d_in[b+11];
    const float* hIn  = L ? hB : hA;
    float*       hOut = L ? hA : hB;

    k_nodelin<<<NNODE, 128, 0, stream>>>(hIn, sc0,sc1,sc2, ln0,ln1,ln2, xself, hlin);
    k_zero   <<<(NNODE*FMID+255)/256, 256, 0, stream>>>(agg, NNODE*FMID);
    k_w2t    <<<(WNUM*64+255)/256, 256, 0, stream>>>(rw2, w2t);
    k_edge   <<<NEDGE/16, 256, 0, stream>>>(pos, ei, hlin, rw0,rb0,rw1,rb1,
                                            w2t, rb2, W3, agg);
    k_gate   <<<(NNODE+255)/256, 256, 0, stream>>>(hIn, xself, agg, CST, hOut);
  }

  float* hFin = hA;  // after layer 1 (L=1 writes hA)
  k_pool_logit<<<NNODE, 128, 0, stream>>>(hFin, pw0,pb0,pw1,pb1,pw2,pb2, logit);
  k_zero      <<<3, 256, 0, stream>>>((float*)umax, 16+16+512);
  k_segmax    <<<NNODE/256, 256, 0, stream>>>(logit, batch, umax);
  k_expsum    <<<NNODE/256, 256, 0, stream>>>(logit, batch, umax, exv, zsum);
  k_pooled    <<<NNODE/256, 256, 0, stream>>>(exv, zsum, batch, hFin, pooled);
  k_out       <<<NGRAPH, 256, 0, stream>>>(pooled, ow, ob, og, obe, (float*)d_out);
}
